// OuterProductMix_53884659696049
// MI455X (gfx1250) — matrix operand order repaired and checked by reading, not compiled, not run on hardware
//
#include <hip/hip_runtime.h>

typedef __attribute__((ext_vector_type(16))) _Float16 v16h;
typedef __attribute__((ext_vector_type(8)))  _Float16 v8h;
typedef __attribute__((ext_vector_type(4)))  _Float16 v4h;
typedef __attribute__((ext_vector_type(8)))  float    v8f;
typedef __attribute__((ext_vector_type(4)))  float    v4f;

#define D_DIM   256
#define K_DIM   65536     // D*D
#define TOKENS  4096
#define LP      264       // padded LDS row stride (halves): 528B -> 4-bank shift/row
#define KSLICES 4         // deterministic split-K factor

// ---------------------------------------------------------------------------
// Pre-pass: W (fp32, 64 MiB) -> f16 (32 MiB) in workspace; L2-resident after.
// ---------------------------------------------------------------------------
__global__ __launch_bounds__(256) void w_to_f16(const float* __restrict__ W,
                                                _Float16* __restrict__ w16,
                                                int n4) {
  int idx    = blockIdx.x * 256 + threadIdx.x;
  int stride = gridDim.x * 256;
  const v4f* src = (const v4f*)W;
  v4h*       dst = (v4h*)w16;
  for (int i = idx; i < n4; i += stride) {
    v4f x = src[i];
    v4h y;
    y[0] = (_Float16)x[0]; y[1] = (_Float16)x[1];
    y[2] = (_Float16)x[2]; y[3] = (_Float16)x[3];
    dst[i] = y;
  }
}

// ---------------------------------------------------------------------------
// Fused bilinear GEMM, split-K partials:
//   part[s,t,o] = sum_{k in slice s} P[t,k] * W16[o,k],  P[t,i*256+j]=h_i*h_j
// Wave tile: 64 tokens x 64 outputs (4 m-frags x 4 n-frags, 128 acc VGPRs);
// A fragments built on the fly and amortized over 4 WMMAs each.
// WG = 4 waves along N (full 256 cols); grid = 64 m-blocks x 4 k-slices
// => 1024 waves.
// ---------------------------------------------------------------------------
__global__ __launch_bounds__(128) void bilinear_wmma(
    const float*    __restrict__ h,
    const _Float16* __restrict__ w16,
    float*          __restrict__ part) {
  __shared__ _Float16 hl[64 * LP];   // 33,792 B

  const int lane   = threadIdx.x & 31;
  const int wid    = threadIdx.x >> 5;     // 0..3
  const int mblk   = blockIdx.x;           // 64-token block
  const int kslice = blockIdx.y;           // 0..3 -> i in [kslice*64, kslice*64+64)
  const int nbase  = wid * 64;             // wave's output-col base

  // ---- stage h tile (64 tokens x 256 f32) into LDS as f16 ----
  {
    const v4f* src = (const v4f*)(h + (size_t)mblk * 64 * D_DIM);
    for (int i = threadIdx.x; i < 64 * D_DIM / 4; i += 128) {
      v4f x = src[i];
      int row = i >> 6;
      int col = (i & 63) << 2;
      v4h y;
      y[0] = (_Float16)x[0]; y[1] = (_Float16)x[1];
      y[2] = (_Float16)x[2]; y[3] = (_Float16)x[3];
      *(v4h*)(&hl[row * LP + col]) = y;
    }
  }
  __syncthreads();

  const int lhalf = lane >> 4;   // 0: K 0..15 ; 1: K 16..31
  const int lrow  = lane & 15;
  const int i0    = kslice * 64;

  // Per-lane B pointers (column o, consecutive K), pre-offset to this k-slice.
  const _Float16* wp[4];
#pragma unroll
  for (int g = 0; g < 4; ++g)
    wp[g] = w16 + (size_t)(nbase + g * 16 + lrow) * K_DIM + (size_t)i0 * D_DIM
                + lhalf * 16;

  // Per-lane A row pointers in LDS (row t = f*16 + lrow).
  const _Float16* arow[4];
#pragma unroll
  for (int f = 0; f < 4; ++f) arow[f] = &hl[(f * 16 + lrow) * LP];

  v8f acc[4][4];
#pragma unroll
  for (int f = 0; f < 4; ++f)
#pragma unroll
    for (int g = 0; g < 4; ++g)
      acc[f][g] = (v8f){0.f, 0.f, 0.f, 0.f, 0.f, 0.f, 0.f, 0.f};

  for (int ii = 0; ii < 64; ++ii) {
    const int i = i0 + ii;
    // per-m-frag scalar h[t,i]; compiler broadcasts via op_sel on v_pk_mul
    _Float16 s[4];
#pragma unroll
    for (int f = 0; f < 4; ++f) s[f] = arow[f][i];

#pragma unroll 2
    for (int jb = 0; jb < 8; ++jb) {
      const int j0 = jb * 32;
      const int kk = ii * D_DIM + j0;   // relative to slice base

      // B fragments: 32 contiguous bytes per lane (2x global_load_b128 each).
      v16h b[4];
#pragma unroll
      for (int g = 0; g < 4; ++g) b[g] = *(const v16h*)(wp[g] + kk);

      // A fragments: s * h[t, j0..j0+31].
      v16h a[4];
#pragma unroll
      for (int f = 0; f < 4; ++f) {
        v8h c0 = *(const v8h*)(arow[f] + j0 + lhalf * 16);
        v8h c1 = *(const v8h*)(arow[f] + j0 + lhalf * 16 + 8);
        v8h sv;
#pragma unroll
        for (int e = 0; e < 8; ++e) sv[e] = s[f];
        v8h p0 = c0 * sv;
        v8h p1 = c1 * sv;
        a[f] = __builtin_shufflevector(p0, p1, 0, 1, 2, 3, 4, 5, 6, 7,
                                       8, 9, 10, 11, 12, 13, 14, 15);
      }

#pragma unroll
      for (int f = 0; f < 4; ++f)
#pragma unroll
        for (int g = 0; g < 4; ++g)
          acc[f][g] = __builtin_amdgcn_wmma_f32_16x16x32_f16(
              false, a[f], false, b[g], (short)0, acc[f][g], false, false);
    }
  }

  // Epilogue: C/D layout -> M = v + 8*lhalf, N = lrow. Store f32 partials.
  float* pbase = part + (size_t)kslice * TOKENS * D_DIM;
#pragma unroll
  for (int g = 0; g < 4; ++g) {
    const int ocol = nbase + g * 16 + lrow;
#pragma unroll
    for (int f = 0; f < 4; ++f) {
      const int rbase = mblk * 64 + f * 16 + lhalf * 8;
#pragma unroll
      for (int v = 0; v < 8; ++v)
        pbase[(size_t)(rbase + v) * D_DIM + ocol] = acc[f][g][v];
    }
  }
}

// ---------------------------------------------------------------------------
// Deterministic split-K reduction + bias add (fixed summation order).
// ---------------------------------------------------------------------------
__global__ __launch_bounds__(256) void reduce_bias(const float* __restrict__ part,
                                                   const float* __restrict__ bias,
                                                   float* __restrict__ out) {
  const int n4 = TOKENS * D_DIM / 4;          // 262,144 v4f
  int idx    = blockIdx.x * 256 + threadIdx.x;
  int stride = gridDim.x * 256;
  const v4f* p = (const v4f*)part;
  v4f*       o = (v4f*)out;
  const int  S = TOKENS * D_DIM / 4;          // v4f stride between slices
  for (int i = idx; i < n4; i += stride) {
    v4f acc = p[i];
    acc = acc + p[i + S];
    acc = acc + p[i + 2 * S];
    acc = acc + p[i + 3 * S];
    v4f bv = *(const v4f*)(bias + ((i * 4) & (D_DIM - 1)));
    o[i] = acc + bv;
  }
}

// ---------------------------------------------------------------------------
extern "C" void kernel_launch(void* const* d_in, const int* in_sizes, int n_in,
                              void* d_out, int out_size, void* d_ws, size_t ws_size,
                              hipStream_t stream) {
  const float* h    = (const float*)d_in[0];   // (4,1024,256) f32
  const float* W    = (const float*)d_in[1];   // (256, 65536) f32
  const float* bias = (const float*)d_in[2];   // (256,) f32
  float*       out  = (float*)d_out;           // (4,1024,256) f32

  _Float16* w16  = (_Float16*)d_ws;                          // 32 MiB
  float*    part = (float*)((char*)d_ws + (size_t)K_DIM * D_DIM * sizeof(_Float16));
                                                             // + 16 MiB partials

  w_to_f16<<<2048, 256, 0, stream>>>(W, w16, K_DIM * D_DIM / 4);
  bilinear_wmma<<<dim3(64, KSLICES), 128, 0, stream>>>(h, w16, part);
  reduce_bias<<<1024, 256, 0, stream>>>(part, bias, out);
}
